// GatedDeltaNet_72859825210077
// MI455X (gfx1250) — matrix operand order repaired and checked by reading, not compiled, not run on hardware
//
#include <hip/hip_runtime.h>

typedef __attribute__((ext_vector_type(16))) __bf16 v16bf;
typedef __attribute__((ext_vector_type(8)))  __bf16 v8bf;
typedef __attribute__((ext_vector_type(4)))  __bf16 v4bf;
typedef __attribute__((ext_vector_type(8)))  float  v8f;
typedef __attribute__((ext_vector_type(4)))  unsigned int u32x4;
typedef __attribute__((ext_vector_type(8)))  int i32x8;
typedef __attribute__((ext_vector_type(4)))  int i32x4;

#define EPS 1e-6f

__device__ __forceinline__ float fast_sigmoid(float x)
{
    return __builtin_amdgcn_rcpf(1.f + __expf(-x));
}

// ---------------- fp32 -> bf16 convert (vectorized, grid-stride) ----------------
__global__ __launch_bounds__(256)
void f32_to_bf16_kernel(const float* __restrict__ in, __bf16* __restrict__ out, long n4)
{
    long i = (long)blockIdx.x * blockDim.x + threadIdx.x;
    long stride = (long)gridDim.x * blockDim.x;
    for (; i < n4; i += stride) {
        float4 v = ((const float4*)in)[i];
        v4bf o;
        o[0] = (__bf16)v.x; o[1] = (__bf16)v.y; o[2] = (__bf16)v.z; o[3] = (__bf16)v.w;
        ((v4bf*)out)[i] = o;
    }
}

// ---------------- Wo_red[h*128+i][d] = sum_g Wo[(3h+g)*128+i][d]  (bf16) ----------------
__global__ __launch_bounds__(256)
void wo_reduce_kernel(const float* __restrict__ Wo, __bf16* __restrict__ out)
{
    int d = blockIdx.x * 256 + threadIdx.x;   // 0..5119
    int r = blockIdx.y;                        // 0..2047
    int h = r >> 7, i = r & 127;
    long base = ((long)(h * 3) * 128 + i) * 5120 + d;
    float s = Wo[base] + Wo[base + (long)128 * 5120] + Wo[base + (long)256 * 5120];
    out[(long)r * 5120 + d] = (__bf16)s;
}

// ---------------- bf16 WMMA GEMM: C[M,N] (f32) = A[M,K] * B[K,N] ----------------
// Double-buffered LDS; A tiles staged by the Tensor Data Mover (TDM) with LDS
// padding (4 dwords per 16 dwords -> 40-element row stride); B tiles register-
// staged and transposed into LDS by the waves. One barrier per K-step.
// MODE 0: store f32; MODE 1: store sigmoid(acc + bias[n])
template <int MODE>
__global__ __launch_bounds__(256)
void gemm_bf16_wmma(const __bf16* __restrict__ A, const __bf16* __restrict__ Bw,
                    float* __restrict__ C, int M, int N, int K,
                    const float* __restrict__ bias)
{
    __shared__ __bf16 As[2][128 * 40];   // row stride 40 elems (80B) via TDM padding
    __shared__ __bf16 Bs[2][128 * 40];   // transposed: Bs[buf][n*40 + k]

    const int tid   = threadIdx.x;
    const int m0    = blockIdx.y * 128;
    const int n0    = blockIdx.x * 128;
    const int wave  = tid >> 5;
    const int lane  = tid & 31;
    const int wm    = wave & 1;   // 0..1  -> 64 rows each
    const int wn    = wave >> 1;  // 0..3  -> 32 cols each
    const int lhalf = lane >> 4;  // 0/1
    const int l15   = lane & 15;

    const int nk = K >> 5;

    // ---- TDM descriptor issue: A tile (128 rows x 32 cols bf16) -> As[buf]
    auto issueA = [&](int kt, int buf) {
        unsigned lds = (unsigned)(unsigned long long)(uintptr_t)(&As[buf][0]);
        unsigned long long ga = (unsigned long long)(uintptr_t)A
                              + ((unsigned long long)((long)m0 * K + (long)kt * 32) << 1);
        u32x4 g0;
        g0[0] = 1u;                                   // count=1 (valid user descriptor)
        g0[1] = lds;                                  // lds_addr
        g0[2] = (unsigned)(ga & 0xFFFFFFFFull);       // global_addr[31:0]
        g0[3] = (unsigned)((ga >> 32) & 0x01FFFFFFu)  // global_addr[56:32]
              | (2u << 30);                           // type = 2 ("image")
        i32x8 g1;
        g1[0] = (1 << 16)      // data_size = 2 bytes
              | (1 << 20)      // pad_enable
              | (3 << 22)      // pad_interval: every 16 dwords (one 32-elem row)
              | (3 << 25);     // pad_amount: 4 dwords -> row stride 80B (40 elems)
        g1[1] = (int)((K & 0xFFFF) << 16);                         // tensor_dim0 lo
        g1[2] = (int)(((unsigned)K >> 16) | ((M & 0xFFFF) << 16)); // dim0 hi | dim1 lo
        g1[3] = (int)(((unsigned)M >> 16) | (32u << 16));          // dim1 hi | tile_dim0=32
        g1[4] = 128;                                               // tile_dim1=128
        g1[5] = (int)K;                                            // tensor_dim0_stride lo32
        g1[6] = 0;
        g1[7] = 0;
        i32x4 gz4 = {0, 0, 0, 0};
        i32x8 gz8 = {0, 0, 0, 0, 0, 0, 0, 0};
        __builtin_amdgcn_tensor_load_to_lds(g0, g1, gz4, gz4, gz8, 0);
    };

    // ---- B tile: global -> regs (issue early), regs -> LDS transposed (late)
    uint4 breg[2];
    auto loadB = [&](int kt) {
#pragma unroll
        for (int s = 0; s < 2; ++s) {
            int idx = tid + s * 256;
            int bk  = idx >> 4;
            int bn  = (idx & 15) * 8;
            breg[s] = *(const uint4*)(Bw + (long)((long)kt * 32 + bk) * N + n0 + bn);
        }
    };
    auto storeB = [&](int buf) {
#pragma unroll
        for (int s = 0; s < 2; ++s) {
            int idx = tid + s * 256;
            int bk  = idx >> 4;
            int bn  = (idx & 15) * 8;
            const __bf16* be = (const __bf16*)&breg[s];
#pragma unroll
            for (int j = 0; j < 8; ++j) Bs[buf][(bn + j) * 40 + bk] = be[j];
        }
    };

    v8f acc[4][2];
#pragma unroll
    for (int i = 0; i < 4; ++i)
#pragma unroll
        for (int j = 0; j < 2; ++j)
#pragma unroll
            for (int r = 0; r < 8; ++r) acc[i][j][r] = 0.f;

    // ---- prologue: stage tile 0
    if (wave == 0) issueA(0, 0);
    loadB(0);
    storeB(0);
    if (wave == 0) __builtin_amdgcn_s_wait_tensorcnt(0);
    __syncthreads();

    for (int kt = 0; kt < nk; ++kt) {
        const int cur = kt & 1;
        const int nxt = cur ^ 1;
        if (kt + 1 < nk) {
            if (wave == 0) issueA(kt + 1, nxt);
            loadB(kt + 1);
        }

        // A fragments: lane = row (l15), lane-half selects K offset {0,8}; elems 8..15 are K+16
        v16bf af[4];
#pragma unroll
        for (int mt = 0; mt < 4; ++mt) {
            int r  = wm * 64 + mt * 16 + l15;
            int ks = lhalf * 8;
            v8bf lo = *(const v8bf*)(&As[cur][r * 40 + ks]);
            v8bf hi = *(const v8bf*)(&As[cur][r * 40 + 16 + ks]);
            af[mt] = __builtin_shufflevector(lo, hi, 0,1,2,3,4,5,6,7,8,9,10,11,12,13,14,15);
        }
        // B fragments: lane = col (l15), lane-half selects K offset {0,8}; elems 8..15 are K+16
        v16bf bfr[2];
#pragma unroll
        for (int nt = 0; nt < 2; ++nt) {
            int c  = wn * 32 + nt * 16 + l15;
            int kb = lhalf * 8;
            v8bf lo = *(const v8bf*)(&Bs[cur][c * 40 + kb]);
            v8bf hi = *(const v8bf*)(&Bs[cur][c * 40 + kb + 16]);
            bfr[nt] = __builtin_shufflevector(lo, hi, 0,1,2,3,4,5,6,7,8,9,10,11,12,13,14,15);
        }
#pragma unroll
        for (int mt = 0; mt < 4; ++mt)
#pragma unroll
            for (int nt = 0; nt < 2; ++nt)
                acc[mt][nt] = __builtin_amdgcn_wmma_f32_16x16x32_bf16(
                    false, af[mt], false, bfr[nt], (short)0, acc[mt][nt], false, false);

        if (kt + 1 < nk) {
            storeB(nxt);
            if (wave == 0) __builtin_amdgcn_s_wait_tensorcnt(0);
        }
        __syncthreads();
    }

    // Epilogue: C layout — VGPR r: lanes0-15 M=r, lanes16-31 M=8+r, N=l15
#pragma unroll
    for (int mt = 0; mt < 4; ++mt)
#pragma unroll
        for (int nt = 0; nt < 2; ++nt) {
            int col = n0 + wn * 32 + nt * 16 + l15;
            float bval = (MODE == 1) ? bias[col] : 0.f;
#pragma unroll
            for (int r = 0; r < 8; ++r) {
                int rowg = m0 + wm * 64 + mt * 16 + lhalf * 8 + r;
                float v = acc[mt][nt][r];
                if (MODE == 1) v = fast_sigmoid(v + bval);
                C[(long)rowg * N + col] = v;
            }
        }
}

// ---------------- beta = sigmoid(x @ Wb + bb), per-row reduction ----------------
__global__ __launch_bounds__(256)
void beta_kernel(const float* __restrict__ x, const float* __restrict__ Wb,
                 const float* __restrict__ bb, float* __restrict__ beta)
{
    __shared__ float red[8][16];
    int m = blockIdx.x;
    int tid = threadIdx.x, lane = tid & 31, wave = tid >> 5;
    float acc[16];
#pragma unroll
    for (int h = 0; h < 16; ++h) acc[h] = 0.f;
    const float* xr = x + (long)m * 5120;
    for (int d = tid; d < 5120; d += 256) {
        float xv = xr[d];
        const float4* wp = (const float4*)(Wb + (long)d * 16);
#pragma unroll
        for (int c = 0; c < 4; ++c) {
            float4 w = wp[c];
            acc[4*c+0] += xv * w.x; acc[4*c+1] += xv * w.y;
            acc[4*c+2] += xv * w.z; acc[4*c+3] += xv * w.w;
        }
    }
#pragma unroll
    for (int h = 0; h < 16; ++h) {
        float v = acc[h];
#pragma unroll
        for (int off = 16; off > 0; off >>= 1) v += __shfl_xor(v, off, 32);
        if (lane == 0) red[wave][h] = v;
    }
    __syncthreads();
    if (tid < 16) {
        float s = 0.f;
#pragma unroll
        for (int w = 0; w < 8; ++w) s += red[w][tid];
        beta[(long)m * 16 + tid] = fast_sigmoid(s + bb[tid]);
    }
}

// ---------------- per-head rmsnorm over DK=128 (in place), one wave per row ----------------
__global__ __launch_bounds__(256)
void rmsnorm128_kernel(float* __restrict__ data, const float* __restrict__ w, int nrows)
{
    int tid = threadIdx.x, lane = tid & 31, wave = tid >> 5;
    int row = blockIdx.x * 8 + wave;
    if (row >= nrows) return;
    float4* p = (float4*)(data + (long)row * 128);
    float4 v = p[lane];
    float ss = v.x*v.x + v.y*v.y + v.z*v.z + v.w*v.w;
#pragma unroll
    for (int off = 16; off > 0; off >>= 1) ss += __shfl_xor(ss, off, 32);
    float scale = rsqrtf(ss * (1.f / 128.f) + EPS);
    float4 wv = ((const float4*)w)[lane];
    v.x *= scale * wv.x; v.y *= scale * wv.y; v.z *= scale * wv.z; v.w *= scale * wv.w;
    p[lane] = v;
}

// ---------------- sequential delta-rule scan ----------------
// 128 blocks: (b, head) x 4 row-splits of the 128x128 state. State rows are
// independent given shared k/q/beta, so S lives entirely in registers
// (16 f32 per thread, 8 threads per row, fully unrolled).
__global__ __launch_bounds__(256)
void scan_kernel(const float* __restrict__ Q, const float* __restrict__ Kn,
                 const float* __restrict__ V, const float* __restrict__ Beta,
                 __bf16* __restrict__ attn)
{
    const int blk   = blockIdx.x;      // 0..127
    const int split = blk & 3;
    const int hb    = blk >> 2;        // 0..31
    const int b     = hb >> 4;
    const int head  = hb & 15;
    const int tid   = threadIdx.x;
    const int rloc  = tid >> 3;        // 0..31 local row
    const int row   = split * 32 + rloc;
    const int sub   = tid & 7;         // 8 lanes per row

    __shared__ float kvec[128], qvec[128], vmean[32];
    __shared__ float bsh;

    float S[16];
#pragma unroll
    for (int j = 0; j < 16; ++j) S[j] = 0.f;

    const long rowQK = (long)b * 2048 * 2048 + (long)head * 128;
    const long rowV  = (long)b * 2048 * 6144 + (long)head * 384;
    const long rowB  = (long)b * 2048 * 16 + head;
    const long rowO  = rowQK + row;

    for (int t = 0; t < 2048; ++t) {
        if (tid < 128) kvec[tid] = Kn[rowQK + (long)t * 2048 + tid];
        else           qvec[tid - 128] = Q[rowQK + (long)t * 2048 + (tid - 128)];
        if (tid < 32) {
            const float* vp = V + rowV + (long)t * 6144 + split * 32 + tid;
            vmean[tid] = (vp[0] + vp[128] + vp[256]) * (1.f / 3.f);
        }
        if (tid == 0) bsh = Beta[rowB + (long)t * 16];
        // pull t+8 rows toward caches while the recurrence grinds
        if (tid == 64 && t + 8 < 2048) {
            __builtin_prefetch(Kn + rowQK + (long)(t + 8) * 2048, 0, 1);
            __builtin_prefetch(Q  + rowQK + (long)(t + 8) * 2048, 0, 1);
            __builtin_prefetch(V  + rowV  + (long)(t + 8) * 6144, 0, 1);
        }
        __syncthreads();

        // v_hat_i = sum_j S[i][j] * k[j]   (columns j = jj*8 + sub)
        float a0 = 0.f, a1 = 0.f;
#pragma unroll
        for (int jj = 0; jj < 16; jj += 2) {
            a0 += S[jj]     * kvec[jj * 8 + sub];
            a1 += S[jj + 1] * kvec[(jj + 1) * 8 + sub];
        }
        float vh = a0 + a1;
        vh += __shfl_xor(vh, 1, 32);
        vh += __shfl_xor(vh, 2, 32);
        vh += __shfl_xor(vh, 4, 32);

        float bt = bsh;
        float bd = bt * (vmean[rloc] - vh);   // beta * delta_i
        float om = 1.f - bt;

        // S = (1-b)S + b*delta_i*k ; o_i = sum_j S_new[i][j]*q[j]
        float o0 = 0.f, o1 = 0.f;
#pragma unroll
        for (int jj = 0; jj < 16; jj += 2) {
            { float kv = kvec[jj * 8 + sub];       float sv = S[jj]     * om + bd * kv; S[jj]     = sv; o0 += sv * qvec[jj * 8 + sub]; }
            { float kv = kvec[(jj + 1) * 8 + sub]; float sv = S[jj + 1] * om + bd * kv; S[jj + 1] = sv; o1 += sv * qvec[(jj + 1) * 8 + sub]; }
        }
        float oi = o0 + o1;
        oi += __shfl_xor(oi, 1, 32);
        oi += __shfl_xor(oi, 2, 32);
        oi += __shfl_xor(oi, 4, 32);
        if (sub == 0) attn[rowO + (long)t * 2048] = (__bf16)oi;
        __syncthreads();
    }
}

// ---------------- final: out = rmsnorm_D(pre) * wout_norm * g ----------------
__global__ __launch_bounds__(256)
void out_epilogue_kernel(const float* __restrict__ pre, const float* __restrict__ g,
                         const float* __restrict__ wn, float* __restrict__ out)
{
    __shared__ float red[8];
    __shared__ float sscale;
    int m = blockIdx.x, tid = threadIdx.x, lane = tid & 31, wave = tid >> 5;
    const float* pr = pre + (long)m * 5120;
    float ss = 0.f;
    for (int d = tid; d < 5120; d += 256) { float v = pr[d]; ss += v * v; }
#pragma unroll
    for (int off = 16; off > 0; off >>= 1) ss += __shfl_xor(ss, off, 32);
    if (lane == 0) red[wave] = ss;
    __syncthreads();
    if (tid == 0) {
        float t = 0.f;
#pragma unroll
        for (int w = 0; w < 8; ++w) t += red[w];
        sscale = rsqrtf(t * (1.f / 5120.f) + EPS);
    }
    __syncthreads();
    float scale = sscale;
    const float* gr = g + (long)m * 5120;
    float* outr = out + (long)m * 5120;
    for (int d = tid; d < 5120; d += 256)
        outr[d] = pr[d] * scale * wn[d] * gr[d];
}

// ---------------- host launcher ----------------
extern "C" void kernel_launch(void* const* d_in, const int* in_sizes, int n_in,
                              void* d_out, int out_size, void* d_ws, size_t ws_size,
                              hipStream_t stream)
{
    (void)in_sizes; (void)n_in; (void)out_size; (void)ws_size;

    const float* x  = (const float*)d_in[0];
    const float* Wq = (const float*)d_in[1];
    const float* Wk = (const float*)d_in[2];
    const float* Wv = (const float*)d_in[3];
    const float* Wb = (const float*)d_in[4];
    const float* bb = (const float*)d_in[5];
    const float* Wg = (const float*)d_in[6];
    const float* bg = (const float*)d_in[7];
    const float* Wo = (const float*)d_in[8];
    const float* wk_norm   = (const float*)d_in[9];
    const float* wout_norm = (const float*)d_in[10];

    const int M = 4096, D = 5120;

    char* ws = (char*)d_ws;
    auto alloc = [&](size_t bytes) -> char* {
        char* p = ws; ws += (bytes + 255) & ~(size_t)255; return p;
    };

    __bf16* xb    = (__bf16*)alloc((size_t)M * D * 2);
    __bf16* Wqb   = (__bf16*)alloc((size_t)D * 2048 * 2);
    __bf16* Wkb   = (__bf16*)alloc((size_t)D * 2048 * 2);
    __bf16* Wvb   = (__bf16*)alloc((size_t)D * 6144 * 2);
    __bf16* Wgb   = (__bf16*)alloc((size_t)D * D * 2);
    __bf16* Worb  = (__bf16*)alloc((size_t)2048 * D * 2);
    float*  qb    = (float*)alloc((size_t)M * 2048 * 4);
    float*  kb    = (float*)alloc((size_t)M * 2048 * 4);
    float*  vb    = (float*)alloc((size_t)M * 6144 * 4);
    float*  gb    = (float*)alloc((size_t)M * D * 4);
    float*  betab = (float*)alloc((size_t)M * 16 * 4);
    __bf16* attnb = (__bf16*)alloc((size_t)M * 2048 * 2);
    float*  preb  = (float*)alloc((size_t)M * D * 4);

    // bf16 conversions
    f32_to_bf16_kernel<<<2048, 256, 0, stream>>>(x,  xb,  (long)M * D / 4);
    f32_to_bf16_kernel<<<2048, 256, 0, stream>>>(Wq, Wqb, (long)D * 2048 / 4);
    f32_to_bf16_kernel<<<2048, 256, 0, stream>>>(Wk, Wkb, (long)D * 2048 / 4);
    f32_to_bf16_kernel<<<2048, 256, 0, stream>>>(Wv, Wvb, (long)D * 6144 / 4);
    f32_to_bf16_kernel<<<2048, 256, 0, stream>>>(Wg, Wgb, (long)D * D / 4);
    wo_reduce_kernel<<<dim3(20, 2048), 256, 0, stream>>>(Wo, Worb);

    // projections (WMMA, TDM-staged A tiles)
    gemm_bf16_wmma<0><<<dim3(16, 32), 256, 0, stream>>>(xb, Wqb, qb, M, 2048, D, nullptr);
    gemm_bf16_wmma<0><<<dim3(16, 32), 256, 0, stream>>>(xb, Wkb, kb, M, 2048, D, nullptr);
    gemm_bf16_wmma<0><<<dim3(48, 32), 256, 0, stream>>>(xb, Wvb, vb, M, 6144, D, nullptr);
    gemm_bf16_wmma<1><<<dim3(40, 32), 256, 0, stream>>>(xb, Wgb, gb, M, D, D, bg);

    beta_kernel<<<M, 256, 0, stream>>>(x, Wb, bb, betab);
    rmsnorm128_kernel<<<M * 16 / 8, 256, 0, stream>>>(kb, wk_norm, M * 16);

    // sequential scan -> attention output (bf16, already group-collapsed to H*DK)
    scan_kernel<<<128, 256, 0, stream>>>(qb, kb, vb, betab, attnb);

    // output projection with group-reduced Wo (WMMA), then rmsnorm * g
    gemm_bf16_wmma<0><<<dim3(40, 32), 256, 0, stream>>>(attnb, Worb, preb, M, D, 2048, nullptr);
    out_epilogue_kernel<<<M, 256, 0, stream>>>(preb, gb, wout_norm, (float*)d_out);
}
